// FlowNet3DDiscriminatorP_1872605741584
// MI455X (gfx1250) — compile-verified, matrix operand order checked by reading, not run on hardware
//
#include <hip/hip_runtime.h>
#include <cstdint>
#include <cstddef>

typedef __attribute__((ext_vector_type(16))) _Float16 v16h;
typedef __attribute__((ext_vector_type(8)))  float    v8f;

#define EPSF 1e-5f
#define CDIV(a,b) (((a)+(b)-1)/(b))
static inline int pad32(int x){ return (x+31)&~31; }

// ---------------- transpose (B,C,N) -> (B,N,C) ----------------
__global__ void k_bcn_to_bnc(const float* __restrict__ src, float* __restrict__ dst,
                             int B, int C, int N) {
  long i = (long)blockIdx.x * blockDim.x + threadIdx.x;
  long tot = (long)B * C * N;
  if (i >= tot) return;
  int b = (int)(i / ((long)C * N));
  int rem = (int)(i - (long)b * C * N);
  int c = rem / N, n = rem % N;
  dst[((long)b * N + n) * C + c] = src[i];
}

// ---------------- farthest point sampling (1 block per batch) ----------------
__global__ void k_fps(const float* __restrict__ xyz, int B, int N, int S,
                      int* __restrict__ out) {
  int b = blockIdx.x;
  const float* X = xyz + (size_t)b * N * 3;
  __shared__ float dmin[4096];
  __shared__ float rv[256];
  __shared__ int   ri[256];
  __shared__ int   s_far;
  int tid = threadIdx.x;
  for (int n = tid; n < N; n += 256) dmin[n] = 1e10f;
  if (tid == 0) s_far = 0;
  __syncthreads();
  for (int it = 0; it < S; ++it) {
    int far = s_far;
    if (tid == 0) out[b * S + it] = far;
    float cx = X[far*3], cy = X[far*3+1], cz = X[far*3+2];
    float bv = -1.f; int bi = 0;
    for (int n = tid; n < N; n += 256) {
      float dx = X[n*3]-cx, dy = X[n*3+1]-cy, dz = X[n*3+2]-cz;
      float d = dx*dx + dy*dy + dz*dz;
      float dm = fminf(dmin[n], d);
      dmin[n] = dm;
      if (dm > bv) { bv = dm; bi = n; }
    }
    rv[tid] = bv; ri[tid] = bi;
    __syncthreads();
    for (int s = 128; s > 0; s >>= 1) {
      if (tid < s) {
        if (rv[tid+s] > rv[tid] || (rv[tid+s] == rv[tid] && ri[tid+s] < ri[tid])) {
          rv[tid] = rv[tid+s]; ri[tid] = ri[tid+s];
        }
      }
      __syncthreads();
    }
    if (tid == 0) s_far = ri[0];
    __syncthreads();
  }
}

__global__ void k_gather_xyz(const float* __restrict__ xyz, const int* __restrict__ fidx,
                             int B, int N, int S, float* __restrict__ nxyz) {
  int i = blockIdx.x * blockDim.x + threadIdx.x;
  if (i >= B * S) return;
  int b = i / S;
  int gi = fidx[i];
  const float* p = xyz + ((size_t)b * N + gi) * 3;
  float* o = nxyz + (size_t)i * 3;
  o[0] = p[0]; o[1] = p[1]; o[2] = p[2];
}

// ---------------- ball query ----------------
__global__ void k_ball(const float* __restrict__ xyz, const float* __restrict__ nxyz,
                       int B, int N, int S, int K, float r2, int* __restrict__ idx) {
  int i = blockIdx.x * blockDim.x + threadIdx.x;
  if (i >= B * S) return;
  int b = i / S;
  const float* q = nxyz + (size_t)i * 3;
  const float* P = xyz + (size_t)b * N * 3;
  float qx = q[0], qy = q[1], qz = q[2];
  int* o = idx + (size_t)i * K;
  int cnt = 0, first = 0; bool havefirst = false;
  for (int n = 0; n < N && cnt < K; ++n) {
    float dx = P[n*3]-qx, dy = P[n*3+1]-qy, dz = P[n*3+2]-qz;
    float d = dx*dx + dy*dy + dz*dz;
    if (d <= r2) {
      if (!havefirst) { first = n; havefirst = true; }
      o[cnt++] = n;
    }
  }
  for (; cnt < K; ++cnt) o[cnt] = first;
}

// ---------------- kNN: one wave per query, lexicographic selection ----------------
__global__ void k_knn(const float* __restrict__ q, const float* __restrict__ kp,
                      int B, int S, int M, int K, int* __restrict__ idx) {
  int gw = (blockIdx.x * blockDim.x + threadIdx.x) >> 5;
  int lane = threadIdx.x & 31;
  if (gw >= B * S) return;
  int b = gw / S, s = gw % S;
  const float* Q = q + ((size_t)b * S + s) * 3;
  const float* P = kp + (size_t)b * M * 3;
  float qx = Q[0], qy = Q[1], qz = Q[2];
  float pd = -1.f; int pi = -1;
  for (int j = 0; j < K; ++j) {
    float bd = 3.0e38f; int bi = 0x7fffffff;
    for (int n = lane; n < M; n += 32) {
      float dx = P[n*3]-qx, dy = P[n*3+1]-qy, dz = P[n*3+2]-qz;
      float d = dx*dx + dy*dy + dz*dz;
      bool after = (d > pd) || (d == pd && n > pi);
      bool better = (d < bd) || (d == bd && n < bi);
      if (after && better) { bd = d; bi = n; }
    }
    for (int off = 16; off > 0; off >>= 1) {
      float od = __shfl_xor(bd, off, 32);
      int   oi = __shfl_xor(bi, off, 32);
      if (od < bd || (od == bd && oi < bi)) { bd = od; bi = oi; }
    }
    if (lane == 0) idx[((size_t)b * S + s) * K + j] = bi;
    pd = bd; pi = bi;
  }
}

// ---------------- grouping kernels -> fp16 padded GEMM operand ----------------
// sa: [xyz[gi]-center (3), feats[gi] (C)] pad to Kp
__global__ void k_group_sa(const float* __restrict__ xyz, const float* __restrict__ feats,
                           const float* __restrict__ nxyz, const int* __restrict__ idx,
                           int B, int Np, int S, int K, int C, int Kp,
                           _Float16* __restrict__ A) {
  long r = (long)blockIdx.x * blockDim.x + threadIdx.x;
  long rows = (long)B * S * K;
  if (r >= rows) return;
  int k = (int)(r % K);
  int s = (int)((r / K) % S);
  int b = (int)(r / ((long)K * S));
  (void)k;
  int gi = idx[r];
  const float* pc = nxyz + ((size_t)b * S + s) * 3;
  const float* pg = xyz + ((size_t)b * Np + gi) * 3;
  const float* fg = feats + ((size_t)b * Np + gi) * C;
  _Float16* Ar = A + (size_t)r * Kp;
  Ar[0] = (_Float16)(pg[0] - pc[0]);
  Ar[1] = (_Float16)(pg[1] - pc[1]);
  Ar[2] = (_Float16)(pg[2] - pc[2]);
  int c = 0;
  for (; c < C; ++c) Ar[3 + c] = (_Float16)fg[c];
  for (c = 3 + C; c < Kp; ++c) Ar[c] = (_Float16)0.f;
}

// flow embedding: [pos_diff (3), f2[gi] (C2), f1[s] (C1)]
__global__ void k_group_fe(const float* __restrict__ p1, const float* __restrict__ p2,
                           const float* __restrict__ f1, const float* __restrict__ f2,
                           const int* __restrict__ idx, int B, int S, int M, int K,
                           int C1, int C2, int Kp, _Float16* __restrict__ A) {
  long r = (long)blockIdx.x * blockDim.x + threadIdx.x;
  long rows = (long)B * S * K;
  if (r >= rows) return;
  int s = (int)((r / K) % S);
  int b = (int)(r / ((long)K * S));
  int gi = idx[r];
  const float* pc = p1 + ((size_t)b * S + s) * 3;
  const float* pg = p2 + ((size_t)b * M + gi) * 3;
  const float* g2 = f2 + ((size_t)b * M + gi) * C2;
  const float* g1 = f1 + ((size_t)b * S + s) * C1;
  _Float16* Ar = A + (size_t)r * Kp;
  Ar[0] = (_Float16)(pg[0] - pc[0]);
  Ar[1] = (_Float16)(pg[1] - pc[1]);
  Ar[2] = (_Float16)(pg[2] - pc[2]);
  int c;
  for (c = 0; c < C2; ++c) Ar[3 + c] = (_Float16)g2[c];
  for (c = 0; c < C1; ++c) Ar[3 + C2 + c] = (_Float16)g1[c];
  for (c = 3 + C2 + C1; c < Kp; ++c) Ar[c] = (_Float16)0.f;
}

// set_upconv mlp1 input: [f2[gi] (C2), pos_diff (3)]
__global__ void k_group_su(const float* __restrict__ p1, const float* __restrict__ p2,
                           const float* __restrict__ f2, const int* __restrict__ idx,
                           int B, int S, int M, int K, int C2, int Kp,
                           _Float16* __restrict__ A) {
  long r = (long)blockIdx.x * blockDim.x + threadIdx.x;
  long rows = (long)B * S * K;
  if (r >= rows) return;
  int s = (int)((r / K) % S);
  int b = (int)(r / ((long)K * S));
  int gi = idx[r];
  const float* pc = p1 + ((size_t)b * S + s) * 3;
  const float* pg = p2 + ((size_t)b * M + gi) * 3;
  const float* g2 = f2 + ((size_t)b * M + gi) * C2;
  _Float16* Ar = A + (size_t)r * Kp;
  int c;
  for (c = 0; c < C2; ++c) Ar[c] = (_Float16)g2[c];
  Ar[C2 + 0] = (_Float16)(pg[0] - pc[0]);
  Ar[C2 + 1] = (_Float16)(pg[1] - pc[1]);
  Ar[C2 + 2] = (_Float16)(pg[2] - pc[2]);
  for (c = C2 + 3; c < Kp; ++c) Ar[c] = (_Float16)0.f;
}

// su1 (empty mlp1): per (b,s,col): max_k over [f2(512), posdiff(3)] then concat feat1(C1)
__global__ void k_su1_pack(const float* __restrict__ p1, const float* __restrict__ p2,
                           const float* __restrict__ f2, const float* __restrict__ f1feat,
                           const int* __restrict__ idx, int B, int S, int M, int K,
                           int C2, int C1, int Kp, _Float16* __restrict__ A) {
  long i = (long)blockIdx.x * blockDim.x + threadIdx.x;
  long tot = (long)B * S * Kp;
  if (i >= tot) return;
  int kp = (int)(i % Kp);
  int bs = (int)(i / Kp);
  int b = bs / S, s = bs % S;
  const int* ix = idx + (size_t)bs * K;
  float v;
  if (kp < C2) {
    v = -3.0e38f;
    for (int j = 0; j < K; ++j) {
      float x = f2[((size_t)b * M + ix[j]) * C2 + kp];
      v = fmaxf(v, x);
    }
  } else if (kp < C2 + 3) {
    int d = kp - C2;
    float pc = p1[((size_t)b * S + s) * 3 + d];
    v = -3.0e38f;
    for (int j = 0; j < K; ++j) {
      float x = p2[((size_t)b * M + ix[j]) * 3 + d] - pc;
      v = fmaxf(v, x);
    }
  } else if (kp < C2 + 3 + C1) {
    v = f1feat[((size_t)bs) * C1 + (kp - C2 - 3)];
  } else {
    v = 0.f;
  }
  A[(size_t)bs * Kp + kp] = (_Float16)v;
}

// ---------------- feature propagation (3-NN interp) ----------------
__global__ void k_fp_weights(const float* __restrict__ q, const float* __restrict__ kp,
                             const int* __restrict__ idx, int B, int N, int M,
                             float* __restrict__ w) {
  int i = blockIdx.x * blockDim.x + threadIdx.x;
  if (i >= B * N) return;
  int b = i / N;
  const float* Q = q + (size_t)i * 3;
  float ww[3], sum = 0.f;
  for (int j = 0; j < 3; ++j) {
    int gi = idx[(size_t)i * 3 + j];
    const float* P = kp + ((size_t)b * M + gi) * 3;
    float dx = P[0]-Q[0], dy = P[1]-Q[1], dz = P[2]-Q[2];
    float d = fmaxf(dx*dx + dy*dy + dz*dz, 1e-10f);
    ww[j] = 1.f / d; sum += ww[j];
  }
  for (int j = 0; j < 3; ++j) w[(size_t)i * 3 + j] = ww[j] / sum;
}

__global__ void k_fp_pack(const float* __restrict__ f2, const float* __restrict__ xyz1,
                          const int* __restrict__ idx, const float* __restrict__ w,
                          int B, int N, int M, int C2, int Kp,
                          _Float16* __restrict__ A) {
  long r = (long)blockIdx.x * blockDim.x + threadIdx.x;
  if (r >= (long)B * N) return;
  int b = (int)(r / N);
  int i0 = idx[r*3], i1 = idx[r*3+1], i2 = idx[r*3+2];
  float w0 = w[r*3], w1 = w[r*3+1], w2 = w[r*3+2];
  const float* F0 = f2 + ((size_t)b * M + i0) * C2;
  const float* F1 = f2 + ((size_t)b * M + i1) * C2;
  const float* F2 = f2 + ((size_t)b * M + i2) * C2;
  _Float16* Ar = A + (size_t)r * Kp;
  int c;
  for (c = 0; c < C2; ++c)
    Ar[c] = (_Float16)(w0 * F0[c] + w1 * F1[c] + w2 * F2[c]);
  const float* X = xyz1 + (size_t)r * 3;
  Ar[C2+0] = (_Float16)X[0]; Ar[C2+1] = (_Float16)X[1]; Ar[C2+2] = (_Float16)X[2];
  for (c = C2 + 3; c < Kp; ++c) Ar[c] = (_Float16)0.f;
}

// ---------------- f32 (M,C) -> f16 (M,Kp) zero padded ----------------
__global__ void k_pack_f16(const float* __restrict__ Y, int M, int C, int Kp,
                           _Float16* __restrict__ A) {
  long i = (long)blockIdx.x * blockDim.x + threadIdx.x;
  long tot = (long)M * Kp;
  if (i >= tot) return;
  int kp = (int)(i % Kp);
  long m = i / Kp;
  A[i] = (kp < C) ? (_Float16)Y[m * C + kp] : (_Float16)0.f;
}

__global__ void k_concat3_pack(const float* __restrict__ X1, int C1,
                               const float* __restrict__ X2, int C2,
                               const float* __restrict__ X3, int C3,
                               int M, int Kp, _Float16* __restrict__ A) {
  long i = (long)blockIdx.x * blockDim.x + threadIdx.x;
  long tot = (long)M * Kp;
  if (i >= tot) return;
  int kp = (int)(i % Kp);
  long m = i / Kp;
  float v = 0.f;
  if (kp < C1) v = X1[m * C1 + kp];
  else if (kp < C1 + C2) v = X2[m * C2 + (kp - C1)];
  else if (kp < C1 + C2 + C3) v = X3[m * C3 + (kp - C1 - C2)];
  A[i] = (_Float16)v;
}

// ---------------- WMMA GEMM: Y[M,Nc] = A[M,Kp] * W[Nc,Kp]^T ----------------
__global__ void k_gemm(const _Float16* __restrict__ A, const _Float16* __restrict__ W,
                       float* __restrict__ Y, int M, int Kp, int Nc) {
  int lane = threadIdx.x & 31;
  int wid = (blockIdx.x * (int)blockDim.x + (int)threadIdx.x) >> 5;
  int tn = Nc >> 4;
  int tiles = (M >> 4) * tn;
  if (wid >= tiles) return;
  int tm = wid / tn, tc = wid % tn;
  int row = (tm << 4) + (lane & 15);
  int col = (tc << 4) + (lane & 15);
  bool hi = lane >= 16;
  const _Float16* arow = A + (size_t)row * Kp;
  const _Float16* wrow = W + (size_t)col * Kp;
  int kofs[8];
#pragma unroll
  for (int v = 0; v < 8; ++v) kofs[v] = ((v < 4) ? 2*v : 8 + 2*v) + (hi ? 8 : 0);
  v8f acc = {};
  union HV { v16h v; uint32_t u[8]; };
  for (int k0 = 0; k0 < Kp; k0 += 32) {
    HV ua, ub;
#pragma unroll
    for (int v = 0; v < 8; ++v) {
      ua.u[v] = *(const uint32_t*)(arow + k0 + kofs[v]);
      ub.u[v] = *(const uint32_t*)(wrow + k0 + kofs[v]);
    }
    acc = __builtin_amdgcn_wmma_f32_16x16x32_f16(false, ua.v, false, ub.v,
                                                 (short)0, acc, false, false);
  }
  int m0 = (tm << 4) + (hi ? 8 : 0);
  int n = (tc << 4) + (lane & 15);
#pragma unroll
  for (int v = 0; v < 8; ++v)
    Y[(size_t)(m0 + v) * Nc + n] = acc[v];
}

// ---------------- batchnorm (per output channel over all rows) + relu ----------------
__global__ void k_zero_f(float* p, long n) {
  long i = (long)blockIdx.x * blockDim.x + threadIdx.x;
  if (i < n) p[i] = 0.f;
}

__global__ void k_colstats(const float* __restrict__ Y, int M, int C, float* st) {
  int tid = threadIdx.x;
  float s0[4] = {0,0,0,0}, s1[4] = {0,0,0,0};
  int rpb = (M + gridDim.x - 1) / gridDim.x;
  int r0 = blockIdx.x * rpb;
  int r1 = min(M, r0 + rpb);
  int nz = (C + (int)blockDim.x - 1) / (int)blockDim.x;
  for (int r = r0; r < r1; ++r) {
    const float* row = Y + (size_t)r * C;
    for (int z = 0; z < nz; ++z) {
      int c = tid + z * blockDim.x;
      if (c < C) { float v = row[c]; s0[z] += v; s1[z] += v * v; }
    }
  }
  for (int z = 0; z < nz; ++z) {
    int c = tid + z * blockDim.x;
    if (c < C) { atomicAdd(&st[c], s0[z]); atomicAdd(&st[C + c], s1[z]); }
  }
}

__global__ void k_bn_relu(float* __restrict__ Y, int M, int C,
                          const float* __restrict__ st,
                          const float* __restrict__ g, const float* __restrict__ b) {
  long i = (long)blockIdx.x * blockDim.x + threadIdx.x;
  long tot = (long)M * C;
  if (i >= tot) return;
  int c = (int)(i % C);
  float mean = st[c] / (float)M;
  float var = st[C + c] / (float)M - mean * mean;
  float v = g[c] * (Y[i] - mean) * rsqrtf(var + EPSF) + b[c];
  Y[i] = fmaxf(v, 0.f);
}

__global__ void k_bias(float* __restrict__ Y, int M, int C, const float* __restrict__ b) {
  long i = (long)blockIdx.x * blockDim.x + threadIdx.x;
  if (i >= (long)M * C) return;
  Y[i] += b[(int)(i % C)];
}

// ---------------- max over neighborhood ----------------
__global__ void k_maxk(const float* __restrict__ Y, int BS, int K, int C,
                       float* __restrict__ out) {
  long i = (long)blockIdx.x * blockDim.x + threadIdx.x;
  if (i >= (long)BS * C) return;
  int c = (int)(i % C);
  long bs = i / C;
  float v = -3.0e38f;
  for (int k = 0; k < K; ++k)
    v = fmaxf(v, Y[((size_t)bs * K + k) * C + c]);
  out[i] = v;
}

// ---------------- output writers ----------------
__global__ void k_out_transpose(const float* __restrict__ Y, int B, int N, int C,
                                float* __restrict__ out) {
  long i = (long)blockIdx.x * blockDim.x + threadIdx.x;
  long tot = (long)B * N * C;
  if (i >= tot) return;
  int c = (int)(i % C);
  int n = (int)((i / C) % N);
  int b = (int)(i / ((long)N * C));
  out[((long)b * C + c) * N + n] = Y[i];
}

__global__ void k_meanrows(const float* __restrict__ X, int B, int Np, int C,
                           float* __restrict__ out) {
  int i = blockIdx.x * blockDim.x + threadIdx.x;
  if (i >= B * C) return;
  int b = i / C, c = i % C;
  float s = 0.f;
  for (int n = 0; n < Np; ++n) s += X[((size_t)b * Np + n) * C + c];
  out[i] = s / (float)Np;
}

// ================= host orchestration =================
struct Lay { const float* W; const float* g; const float* b; int cin; int cout; };

static void run_mlp(hipStream_t st, _Float16* Ain, _Float16* Apong, float* Y,
                    _Float16* W16, float* stats, int M, int Kp0,
                    const Lay* L, int nl) {
  _Float16* A = Ain; _Float16* Ao = Apong;
  int Kp = Kp0;
  for (int i = 0; i < nl; ++i) {
    int cout = L[i].cout;
    { long n = (long)cout * Kp;
      k_pack_f16<<<CDIV(n,256),256,0,st>>>(L[i].W, cout, L[i].cin, Kp, W16); }
    { int tiles = (M/16)*(cout/16);
      k_gemm<<<CDIV(tiles,8),256,0,st>>>(A, W16, Y, M, Kp, cout); }
    k_zero_f<<<CDIV(2*cout,256),256,0,st>>>(stats, 2L*cout);
    k_colstats<<<256,256,0,st>>>(Y, M, cout, stats);
    { long n = (long)M*cout;
      k_bn_relu<<<CDIV(n,256),256,0,st>>>(Y, M, cout, stats, L[i].g, L[i].b); }
    if (i + 1 < nl) {
      Kp = pad32(cout);
      long n = (long)M*Kp;
      k_pack_f16<<<CDIV(n,256),256,0,st>>>(Y, M, cout, Kp, Ao);
      _Float16* t = A; A = Ao; Ao = t;
    }
  }
}

static void sa_stage(hipStream_t st, const float* xyz, const float* feats, int Np, int C,
                     int S, float radius, int K, const Lay* L, int nl,
                     float* out_xyz, float* out_feat,
                     int* fpsidx, int* idxbuf, _Float16* A16a, _Float16* A16b,
                     float* Y, _Float16* W16, float* stats) {
  const int B = 8;
  k_fps<<<B,256,0,st>>>(xyz, B, Np, S, fpsidx);
  k_gather_xyz<<<CDIV(B*S,256),256,0,st>>>(xyz, fpsidx, B, Np, S, out_xyz);
  k_ball<<<CDIV(B*S,256),256,0,st>>>(xyz, out_xyz, B, Np, S, K, radius*radius, idxbuf);
  int Kp = pad32(3 + C);
  long rows = (long)B * S * K;
  k_group_sa<<<CDIV(rows,256),256,0,st>>>(xyz, feats, out_xyz, idxbuf, B, Np, S, K, C, Kp, A16a);
  run_mlp(st, A16a, A16b, Y, W16, stats, (int)rows, Kp, L, nl);
  int cl = L[nl-1].cout;
  long n = (long)B * S * cl;
  k_maxk<<<CDIV(n,256),256,0,st>>>(Y, B*S, K, cl, out_feat);
}

extern "C" void kernel_launch(void* const* d_in, const int* in_sizes, int n_in,
                              void* d_out, int out_size, void* d_ws, size_t ws_size,
                              hipStream_t stream) {
  (void)in_sizes; (void)n_in; (void)out_size; (void)ws_size;
  const int B = 8, N = 4096;
  hipStream_t st = stream;
  auto F = [&](int i){ return (const float*)d_in[i]; };
  const float* pc1 = F(0);
  const float* pc2 = F(1);

  // params: tree-flatten order (sorted dict keys; list order; tuple order)
  Lay conv1L[1] = {{F(2),F(3),F(4),256,128}};
  const float* conv2W = F(5); const float* conv2b = F(6);
  Lay feL[3]   = {{F(7),F(8),F(9),259,128},{F(10),F(11),F(12),128,128},{F(13),F(14),F(15),128,128}};
  Lay fpL[2]   = {{F(16),F(17),F(18),259,256},{F(19),F(20),F(21),256,256}};
  Lay sa1L[3]  = {{F(22),F(23),F(24),6,32},{F(25),F(26),F(27),32,32},{F(28),F(29),F(30),32,64}};
  Lay sa2L[3]  = {{F(31),F(32),F(33),67,64},{F(34),F(35),F(36),64,64},{F(37),F(38),F(39),64,128}};
  Lay sa3L[3]  = {{F(40),F(41),F(42),131,128},{F(43),F(44),F(45),128,128},{F(46),F(47),F(48),128,256}};
  Lay sa4L[3]  = {{F(49),F(50),F(51),259,256},{F(52),F(53),F(54),256,256},{F(55),F(56),F(57),256,512}};
  Lay su1L[2]  = {{F(58),F(59),F(60),771,256},{F(61),F(62),F(63),256,256}};
  Lay su2m1[3] = {{F(64),F(65),F(66),259,128},{F(67),F(68),F(69),128,128},{F(70),F(71),F(72),128,256}};
  Lay su2m2[1] = {{F(73),F(74),F(75),512,256}};
  Lay su3m1[3] = {{F(76),F(77),F(78),259,128},{F(79),F(80),F(81),128,128},{F(82),F(83),F(84),128,256}};
  Lay su3m2[1] = {{F(85),F(86),F(87),320,256}};

  // -------- workspace bump allocator --------
  size_t off = 0;
  char* base = (char*)d_ws;
  auto AL = [&](size_t bytes) -> void* {
    void* p = base + off;
    off = (off + bytes + 255) & ~(size_t)255;
    return p;
  };
  float* xyz1   = (float*)AL(98304*4);
  float* xyz2   = (float*)AL(98304*4);
  float* l1_p1  = (float*)AL(24576*4);
  float* l1_f1  = (float*)AL(524288*4);
  float* l2_p1  = (float*)AL(6144*4);
  float* l2_f1  = (float*)AL(262144*4);
  float* l1_p2  = (float*)AL(24576*4);
  float* l1_f2  = (float*)AL(524288*4);
  float* l2_p2  = (float*)AL(6144*4);
  float* l2_f2  = (float*)AL(262144*4);
  float* l2_f1n = (float*)AL(262144*4);
  float* l3_p1  = (float*)AL(1536*4);
  float* l3_f1  = (float*)AL(131072*4);
  float* l4_p1  = (float*)AL(384*4);
  float* l4_f1  = (float*)AL(65536*4);
  float* l3_fn  = (float*)AL(131072*4);
  float* l2_fn  = (float*)AL(524288*4);
  float* l1_fn  = (float*)AL(2097152*4);
  float* l0_fn  = (float*)AL(8388608*4);
  float* maxbuf = (float*)AL(2097152*4);
  float* knnw   = (float*)AL(98304*4);
  float* stats  = (float*)AL(1024*4);
  float* sc1    = (float*)AL((size_t)4194304*4);
  int*   fpsidx = (int*)AL(8192*4);
  int*   idxbuf = (int*)AL(131072*4);
  _Float16* W16  = (_Float16*)AL((size_t)409600*2);
  _Float16* A16a = (_Float16*)AL((size_t)37748736*2);
  _Float16* A16b = (_Float16*)AL((size_t)16777216*2);
  float*    Y    = (float*)AL((size_t)16777216*4);

  // -------- input transposes (B,3,N)->(B,N,3) --------
  k_bcn_to_bnc<<<CDIV((long)B*3*N,256),256,0,st>>>(pc1, xyz1, B, 3, N);
  k_bcn_to_bnc<<<CDIV((long)B*3*N,256),256,0,st>>>(pc2, xyz2, B, 3, N);

  // -------- set abstraction levels 1 & 2 for both clouds --------
  sa_stage(st, xyz1, xyz1, N, 3, 1024, 0.5f, 16, sa1L, 3, l1_p1, l1_f1,
           fpsidx, idxbuf, A16a, A16b, Y, W16, stats);
  sa_stage(st, l1_p1, l1_f1, 1024, 64, 256, 1.0f, 16, sa2L, 3, l2_p1, l2_f1,
           fpsidx, idxbuf, A16a, A16b, Y, W16, stats);
  sa_stage(st, xyz2, xyz2, N, 3, 1024, 0.5f, 16, sa1L, 3, l1_p2, l1_f2,
           fpsidx, idxbuf, A16a, A16b, Y, W16, stats);
  sa_stage(st, l1_p2, l1_f2, 1024, 64, 256, 1.0f, 16, sa2L, 3, l2_p2, l2_f2,
           fpsidx, idxbuf, A16a, A16b, Y, W16, stats);

  // -------- flow embedding (knn 64) --------
  {
    int S = 256, M = 256, K = 64;
    k_knn<<<CDIV(B*S*32,256),256,0,st>>>(l2_p1, l2_p2, B, S, M, K, idxbuf);
    long rows = (long)B*S*K;  // 131072
    int Kp = pad32(3+128+128); // 288
    k_group_fe<<<CDIV(rows,256),256,0,st>>>(l2_p1, l2_p2, l2_f1, l2_f2, idxbuf,
                                            B, S, M, K, 128, 128, Kp, A16a);
    run_mlp(st, A16a, A16b, Y, W16, stats, (int)rows, Kp, feL, 3);
    long n = (long)B*S*128;
    k_maxk<<<CDIV(n,256),256,0,st>>>(Y, B*S, K, 128, l2_f1n);
  }

  // -------- sa3, sa4 --------
  sa_stage(st, l2_p1, l2_f1n, 256, 128, 64, 2.0f, 8, sa3L, 3, l3_p1, l3_f1,
           fpsidx, idxbuf, A16a, A16b, Y, W16, stats);
  sa_stage(st, l3_p1, l3_f1, 64, 256, 16, 4.0f, 8, sa4L, 3, l4_p1, l4_f1,
           fpsidx, idxbuf, A16a, A16b, Y, W16, stats);

  // -------- su1: l3 <- l4 (empty mlp1) --------
  {
    int S = 64, M = 16, K = 8, Kp = pad32(771); // 800
    k_knn<<<CDIV(B*S*32,256),256,0,st>>>(l3_p1, l4_p1, B, S, M, K, idxbuf);
    long tot = (long)B*S*Kp;
    k_su1_pack<<<CDIV(tot,256),256,0,st>>>(l3_p1, l4_p1, l4_f1, l3_f1, idxbuf,
                                           B, S, M, K, 512, 256, Kp, A16a);
    run_mlp(st, A16a, A16b, Y, W16, stats, B*S, Kp, su1L, 2);
    hipMemcpyAsync(l3_fn, Y, (size_t)B*S*256*4, hipMemcpyDeviceToDevice, st);
  }

  // -------- su2: l2 <- l3 --------
  {
    int S = 256, M = 64, K = 8, Kp = pad32(259); // 288
    k_knn<<<CDIV(B*S*32,256),256,0,st>>>(l2_p1, l3_p1, B, S, M, K, idxbuf);
    long rows = (long)B*S*K; // 16384
    k_group_su<<<CDIV(rows,256),256,0,st>>>(l2_p1, l3_p1, l3_fn, idxbuf,
                                            B, S, M, K, 256, Kp, A16a);
    run_mlp(st, A16a, A16b, Y, W16, stats, (int)rows, Kp, su2m1, 3);
    long n = (long)B*S*256;
    k_maxk<<<CDIV(n,256),256,0,st>>>(Y, B*S, K, 256, maxbuf);
    int Kp2 = pad32(512); // 512
    long tot = (long)B*S*Kp2;
    k_concat3_pack<<<CDIV(tot,256),256,0,st>>>(maxbuf, 256, l2_f1, 128, l2_f1n, 128,
                                               B*S, Kp2, A16a);
    run_mlp(st, A16a, A16b, Y, W16, stats, B*S, Kp2, su2m2, 1);
    hipMemcpyAsync(l2_fn, Y, (size_t)B*S*256*4, hipMemcpyDeviceToDevice, st);
  }

  // -------- su3: l1 <- l2 --------
  {
    int S = 1024, M = 256, K = 8, Kp = pad32(259); // 288
    k_knn<<<CDIV(B*S*32,256),256,0,st>>>(l1_p1, l2_p1, B, S, M, K, idxbuf);
    long rows = (long)B*S*K; // 65536
    k_group_su<<<CDIV(rows,256),256,0,st>>>(l1_p1, l2_p1, l2_fn, idxbuf,
                                            B, S, M, K, 256, Kp, A16a);
    run_mlp(st, A16a, A16b, Y, W16, stats, (int)rows, Kp, su3m1, 3);
    long n = (long)B*S*256;
    k_maxk<<<CDIV(n,256),256,0,st>>>(Y, B*S, K, 256, maxbuf);
    int Kp2 = pad32(320); // 320
    long tot = (long)B*S*Kp2;
    k_concat3_pack<<<CDIV(tot,256),256,0,st>>>(maxbuf, 256, l1_f1, 64, (const float*)nullptr, 0,
                                               B*S, Kp2, A16a);
    run_mlp(st, A16a, A16b, Y, W16, stats, B*S, Kp2, su3m2, 1);
    hipMemcpyAsync(l1_fn, Y, (size_t)B*S*256*4, hipMemcpyDeviceToDevice, st);
  }

  // -------- feature propagation to full resolution --------
  {
    int M = 1024, Kp = pad32(259); // 288
    k_knn<<<CDIV(B*N*32,256),256,0,st>>>(xyz1, l1_p1, B, N, M, 3, idxbuf);
    k_fp_weights<<<CDIV(B*N,256),256,0,st>>>(xyz1, l1_p1, idxbuf, B, N, M, knnw);
    k_fp_pack<<<CDIV((long)B*N,256),256,0,st>>>(l1_fn, xyz1, idxbuf, knnw,
                                                B, N, M, 256, Kp, A16a);
    run_mlp(st, A16a, A16b, Y, W16, stats, B*N, Kp, fpL, 2);
    hipMemcpyAsync(l0_fn, Y, (size_t)B*N*256*4, hipMemcpyDeviceToDevice, st);
  }

  // -------- conv1 (BN+relu) + conv2 (bias only) --------
  {
    int M = B*N; // 32768
    long tot = (long)M*256;
    k_pack_f16<<<CDIV(tot,256),256,0,st>>>(l0_fn, M, 256, 256, A16a);
    run_mlp(st, A16a, A16b, Y, W16, stats, M, 256, conv1L, 1);
    tot = (long)M*128;
    k_pack_f16<<<CDIV(tot,256),256,0,st>>>(Y, M, 128, 128, A16b);
    k_pack_f16<<<CDIV((long)128*128,256),256,0,st>>>(conv2W, 128, 128, 128, W16);
    { int tiles = (M/16)*(128/16);
      k_gemm<<<CDIV(tiles,8),256,0,st>>>(A16b, W16, sc1, M, 128, 128); }
    k_bias<<<CDIV((long)M*128,256),256,0,st>>>(sc1, M, 128, conv2b);
    k_out_transpose<<<CDIV((long)B*N*128,256),256,0,st>>>(sc1, B, N, 128, (float*)d_out);
  }

  // -------- hidden means --------
  float* hid = (float*)d_out + (size_t)B*128*N;
  k_meanrows<<<CDIV(B*256,256),256,0,st>>>(l0_fn, B, 4096, 256, hid); hid += B*256;
  k_meanrows<<<CDIV(B*256,256),256,0,st>>>(l1_fn, B, 1024, 256, hid); hid += B*256;
  k_meanrows<<<CDIV(B*256,256),256,0,st>>>(l2_fn, B, 256, 256, hid);  hid += B*256;
  k_meanrows<<<CDIV(B*256,256),256,0,st>>>(l3_fn, B, 64, 256, hid);   hid += B*256;
  k_meanrows<<<CDIV(B*512,256),256,0,st>>>(l4_f1, B, 16, 512, hid);   hid += B*512;
  k_meanrows<<<CDIV(B*256,256),256,0,st>>>(l3_f1, B, 64, 256, hid);   hid += B*256;
  k_meanrows<<<CDIV(B*128,256),256,0,st>>>(l2_f2, B, 256, 128, hid);  hid += B*128;
  k_meanrows<<<CDIV(B*64,256),256,0,st>>>(l1_f2, B, 1024, 64, hid);
}
